// OSRSImitationModel_39101382262800
// MI455X (gfx1250) — compile-verified, matrix operand order checked by reading, not run on hardware
//
// CDNA5 (gfx1250) implementation of the OSRS imitation model forward pass.
// All GEMMs run on v_wmma_f32_16x16x32_f16 (f16 in, f32 accum). Compute-bound
// workload (~260 GFLOP vs ~75MB HBM traffic), so everything funnels through
// one generic WMMA GEMM kernel: D = act(A[M,K] @ W[N,K]^T + bias + addend).
// Each wave computes a 32x64 output tile (8 WMMAs per K-step, 2 A-frags +
// 4 B-frags) to keep the matrix pipes, not the address ALUs, on the
// critical path. Out-of-range output columns are handled by clamping the
// weight-row index (no exec divergence inside the WMMA loop).
#include <hip/hip_runtime.h>
#include <hip/hip_bf16.h>

typedef _Float16 half_t;
typedef __attribute__((ext_vector_type(16))) _Float16 v16h;
typedef __attribute__((ext_vector_type(8)))  _Float16 v8h;
typedef __attribute__((ext_vector_type(8)))  float    v8f;

#define Bsz  8192
#define Tseq 10
#define Fin  128
#define Hn   256
#define HDn  128
#define Tact 15
#define NCnt 16

static const long Mseq = (long)Bsz * Tseq;   // 81920
static const long Mdec = (long)Bsz * Tact;   // 122880

// ---------------------------------------------------------------------------
// Fragment loader: 16-bit A/B fragment per ISA layout.
// Lane L (0-15): row = L, slots 0..7 = K kb..kb+7, slots 8..15 = K 16+kb..23+kb
// with kb = 0 for lanes 0-15 and kb = 8 for lanes 16-31. Caller passes
// p = row_base + k0 + kb; the high half lives 16 elements further.
// ---------------------------------------------------------------------------
static __device__ inline v16h load_frag(const half_t* __restrict__ p) {
  v8h lo = *(const v8h*)(p);
  v8h hi = *(const v8h*)(p + 16);
  return __builtin_shufflevector(lo, hi, 0,1,2,3,4,5,6,7,8,9,10,11,12,13,14,15);
}

// ---------------------------------------------------------------------------
// Generic WMMA GEMM: out = act(A[M,K] @ W[N,K]^T + bias[N] + Add[M,N])
//   A    : f16, row stride lda, base offset abase (elements); M % 32 == 0
//   W    : f16, (N,K) row-major  (torch weight layout = column-major B operand)
//   Add  : optional f16 addend, row stride ldadd, base addbase
//   o16  : optional f16 output (row stride ld16), o32: optional f32 output
// One wave per 32x64 output tile, K stepped by 32. act: 0=none 1=relu 2=sigm
// ---------------------------------------------------------------------------
__global__ __launch_bounds__(256)
void k_gemm(const half_t* __restrict__ A, long abase, int lda,
            const half_t* __restrict__ W,
            const float*  __restrict__ bias,
            const half_t* __restrict__ Add, long addbase, int ldadd,
            half_t* __restrict__ o16, long o16b, int ld16,
            float*  __restrict__ o32, long o32b, int ld32,
            int N, int K, int act, int ntN, long ntiles)
{
  const int wave = threadIdx.x >> 5;
  const int lane = threadIdx.x & 31;
  const long tile = (long)blockIdx.x * 8 + wave;
  if (tile >= ntiles) return;
  const int tm = (int)(tile / ntN);
  const int tn = (int)(tile % ntN);

  const int l15 = lane & 15;
  const int kb  = (lane >> 4) << 3;          // 0 or 8

  // A rows for the two 16-row subtiles of this wave's 32-row stripe.
  const half_t* __restrict__ arow0 = A + abase + (long)(tm * 32 + l15) * lda + kb;
  const half_t* __restrict__ arow1 = arow0 + (long)16 * lda;

  // W rows (output columns) for the four 16-col subtiles; clamp OOB lanes to
  // a valid row (their products land in columns that are never stored).
  const half_t* __restrict__ wr[4];
#pragma unroll
  for (int j = 0; j < 4; ++j) {
    int n = tn * 64 + j * 16 + l15;
    if (n > N - 1) n = N - 1;
    wr[j] = W + (long)n * K + kb;
    __builtin_prefetch(wr[j], 0, 1);       // global_prefetch_b8 (L2-warm hint)
  }

  v8f acc[2][4] = {};
  for (int k0 = 0; k0 < K; k0 += 32) {
    v16h a0 = load_frag(arow0 + k0);
    v16h a1 = load_frag(arow1 + k0);
#pragma unroll
    for (int j = 0; j < 4; ++j) {
      v16h bf = load_frag(wr[j] + k0);
      acc[0][j] = __builtin_amdgcn_wmma_f32_16x16x32_f16(
          false, a0, false, bf, (short)0, acc[0][j], false, false);
      acc[1][j] = __builtin_amdgcn_wmma_f32_16x16x32_f16(
          false, a1, false, bf, (short)0, acc[1][j], false, false);
    }
  }

  // Epilogue. C/D layout: VGPR r holds M = r (lanes 0-15) / M = r+8 (16-31).
  const int mbase = tm * 32 + ((lane >> 4) << 3);
#pragma unroll
  for (int j = 0; j < 4; ++j) {
    const int n = tn * 64 + j * 16 + l15;
    if (n >= N) continue;
    const float bv = bias ? bias[n] : 0.0f;
#pragma unroll
    for (int i = 0; i < 2; ++i) {
#pragma unroll
      for (int r = 0; r < 8; ++r) {
        const int m = mbase + i * 16 + r;
        float v = acc[i][j][r] + bv;
        if (Add) v += (float)Add[addbase + (long)m * ldadd + n];
        if (act == 1)      v = fmaxf(v, 0.0f);
        else if (act == 2) v = 1.0f / (1.0f + __expf(-v));
        if (o16) o16[o16b + (long)m * ld16 + n] = (half_t)v;
        if (o32) o32[o32b + (long)m * ld32 + n] = v;
      }
    }
  }
}

// ----------------------------- pointwise kernels ---------------------------
__global__ void k_f32_to_f16(const float* __restrict__ s, half_t* __restrict__ d, long n) {
  long i = (long)blockIdx.x * blockDim.x + threadIdx.x;
  if (i < n) d[i] = (half_t)s[i];
}
__global__ void k_f32_to_f16_2d(const float* __restrict__ s, half_t* __restrict__ d,
                                int rows, int cols, int sld) {
  long i = (long)blockIdx.x * blockDim.x + threadIdx.x;
  if (i >= (long)rows * cols) return;
  int r = (int)(i / cols), c = (int)(i % cols);
  d[i] = (half_t)s[(long)r * sld + c];
}
__global__ void k_addvec(const float* a, const float* b, float* o, int n) {
  int i = blockIdx.x * blockDim.x + threadIdx.x;
  if (i < n) o[i] = a[i] + b[i];
}
__global__ void k_zero_h16(half_t* p, long n) {
  long i = (long)blockIdx.x * blockDim.x + threadIdx.x;
  if (i < n) p[i] = (half_t)0.0f;
}
__global__ void k_fill_f32(float* p, float v, long n) {
  long i = (long)blockIdx.x * blockDim.x + threadIdx.x;
  if (i < n) p[i] = v;
}
// per-timestep decoder input bias: (t/Tact) * dec_wih0[:,256]
__global__ void k_dec_bias(const float* __restrict__ wih0, float* __restrict__ out) {
  int i = blockIdx.x * blockDim.x + threadIdx.x;
  if (i >= Tact * 4 * HDn) return;
  int t = i / (4 * HDn), n = i % (4 * HDn);
  out[i] = ((float)t / (float)Tact) * wih0[(long)n * (Hn + 4) + Hn];
}
// gates G[b, 4H] (i,f,g,o) -> update c (f32), h (f16), optional h sequence copy
__global__ void k_lstm_cell(const half_t* __restrict__ G, float* __restrict__ c,
                            half_t* __restrict__ h, half_t* __restrict__ hseq,
                            long hseqb, int hseqld, int Bn, int Hc) {
  long idx = (long)blockIdx.x * blockDim.x + threadIdx.x;
  if (idx >= (long)Bn * Hc) return;
  int b = (int)(idx / Hc), j = (int)(idx % Hc);
  const half_t* g = G + (long)b * 4 * Hc;
  float gi = (float)g[j], gf = (float)g[Hc + j];
  float gg = (float)g[2 * Hc + j], go = (float)g[3 * Hc + j];
  float ig = 1.0f / (1.0f + __expf(-gi));
  float fg = 1.0f / (1.0f + __expf(-gf));
  float og = 1.0f / (1.0f + __expf(-go));
  float cn = fg * c[idx] + ig * tanhf(gg);
  c[idx] = cn;
  float hn = og * tanhf(cn);
  h[idx] = (half_t)hn;
  if (hseq) hseq[hseqb + (long)b * hseqld + j] = (half_t)hn;
}
// decoder init: h1=c1=h2=c2=context[:, :128]
__global__ void k_dec_init(const half_t* __restrict__ ctx, half_t* h1, half_t* h2,
                           float* c1, float* c2) {
  long idx = (long)blockIdx.x * blockDim.x + threadIdx.x;
  if (idx >= (long)Bsz * HDn) return;
  int b = (int)(idx / HDn), j = (int)(idx % HDn);
  half_t v = ctx[(long)b * Hn + j];
  h1[idx] = v; h2[idx] = v;
  float f = (float)v; c1[idx] = f; c2[idx] = f;
}
__global__ void k_softmax16(const float* __restrict__ lg, float* __restrict__ out, int Bn) {
  int b = blockIdx.x * blockDim.x + threadIdx.x;
  if (b >= Bn) return;
  const float* r = lg + (long)b * NCnt;
  float mx = r[0];
#pragma unroll
  for (int i = 1; i < NCnt; ++i) mx = fmaxf(mx, r[i]);
  float e[NCnt], s = 0.0f;
#pragma unroll
  for (int i = 0; i < NCnt; ++i) { e[i] = __expf(r[i] - mx); s += e[i]; }
  float inv = 1.0f / s;
#pragma unroll
  for (int i = 0; i < NCnt; ++i) out[(long)b * NCnt + i] = e[i] * inv;
}

// ----------------------------- host driver --------------------------------
enum {
  IN_X = 0,
  P_ENC_W1, P_ENC_B1, P_ENC_W2, P_ENC_B2,
  P_LSTM_WIH0, P_LSTM_WHH0, P_LSTM_BIH0, P_LSTM_BHH0,
  P_LSTM_WIH1, P_LSTM_WHH1, P_LSTM_BIH1, P_LSTM_BHH1,
  P_CNT_W1, P_CNT_B1, P_CNT_W2, P_CNT_B2,
  P_DEC_WIH0, P_DEC_WHH0, P_DEC_BIH0, P_DEC_BHH0,
  P_DEC_WIH1, P_DEC_WHH1, P_DEC_BIH1, P_DEC_BHH1,
  P_TYPE_W1, P_TYPE_B1, P_TYPE_W2, P_TYPE_B2,
  P_MPOS_W1, P_MPOS_B1, P_MPOS_W2, P_MPOS_B2,
  P_CLK_W1, P_CLK_B1, P_CLK_W2, P_CLK_B2,
  P_KEY_W1, P_KEY_B1, P_KEY_W2, P_KEY_B2,
  P_SCR_W1, P_SCR_B1, P_SCR_W2, P_SCR_B2,
  P_TIM_W1, P_TIM_B1, P_TIM_W2, P_TIM_B2,
  P_CONF_W1, P_CONF_B1, P_CONF_W2, P_CONF_B2,
  IN_MAXACT
};

static inline unsigned blks(long n, int t = 256) { return (unsigned)((n + t - 1) / t); }

struct Bump {
  char* base; size_t used;
  void* take(size_t bytes) {
    size_t a = (used + 255) & ~(size_t)255;
    used = a + bytes;
    return base + a;
  }
  half_t* h16(long elems) { return (half_t*)take((size_t)elems * sizeof(half_t)); }
  float*  f32(long elems) { return (float*)take((size_t)elems * sizeof(float)); }
};

static void gemm(hipStream_t st,
                 const half_t* A, long abase, int lda, const half_t* W,
                 const float* bias, const half_t* Add, long addbase, int ldadd,
                 half_t* o16, long o16b, int ld16,
                 float* o32, long o32b, int ld32,
                 int M, int N, int K, int act) {
  int ntM = M / 32;                 // 32-row stripes (all M here are %32==0)
  int ntN = (N + 63) / 64;          // 64-col strips
  long ntiles = (long)ntM * ntN;
  unsigned nblk = (unsigned)((ntiles + 7) / 8);
  k_gemm<<<nblk, 256, 0, st>>>(A, abase, lda, W, bias, Add, addbase, ldadd,
                               o16, o16b, ld16, o32, o32b, ld32,
                               N, K, act, ntN, ntiles);
}

extern "C" void kernel_launch(void* const* d_in, const int* in_sizes, int n_in,
                              void* d_out, int out_size, void* d_ws, size_t ws_size,
                              hipStream_t stream) {
  (void)in_sizes; (void)n_in; (void)out_size; (void)ws_size;
  float* dout = (float*)d_out;
  Bump ws{(char*)d_ws, 0};

  auto F = [&](int i) { return (const float*)d_in[i]; };

  // ---- f16 weight staging -------------------------------------------------
  auto cvt = [&](int idx, long n) {
    half_t* dst = ws.h16(n);
    k_f32_to_f16<<<blks(n), 256, 0, stream>>>(F(idx), dst, n);
    return dst;
  };
  half_t* w_enc1 = cvt(P_ENC_W1, 256L * Fin);
  half_t* w_enc2 = cvt(P_ENC_W2, 256L * 256);
  half_t* w_ih0  = cvt(P_LSTM_WIH0, 4L * Hn * 256);
  half_t* w_hh0  = cvt(P_LSTM_WHH0, 4L * Hn * Hn);
  half_t* w_ih1  = cvt(P_LSTM_WIH1, 4L * Hn * Hn);
  half_t* w_hh1  = cvt(P_LSTM_WHH1, 4L * Hn * Hn);
  half_t* w_cnt1 = cvt(P_CNT_W1, 128L * Hn);
  half_t* w_cnt2 = cvt(P_CNT_W2, (long)NCnt * 128);
  half_t* w_dhh0 = cvt(P_DEC_WHH0, 4L * HDn * HDn);
  half_t* w_dih1 = cvt(P_DEC_WIH1, 4L * HDn * HDn);
  half_t* w_dhh1 = cvt(P_DEC_WHH1, 4L * HDn * HDn);
  // dec_wih0 is (512, 260); only the first 256 (context) columns feed a GEMM,
  // column 256 is the scalar position input folded into a per-t bias.
  half_t* w_dc = ws.h16(4L * HDn * Hn);
  k_f32_to_f16_2d<<<blks(4L * HDn * Hn), 256, 0, stream>>>(
      F(P_DEC_WIH0), w_dc, 4 * HDn, Hn, Hn + 4);

  // ---- fused biases -------------------------------------------------------
  float* b_l0 = ws.f32(4 * Hn);
  float* b_l1 = ws.f32(4 * Hn);
  float* b_d0 = ws.f32(4 * HDn);
  float* b_d1 = ws.f32(4 * HDn);
  k_addvec<<<blks(4 * Hn), 256, 0, stream>>>(F(P_LSTM_BIH0), F(P_LSTM_BHH0), b_l0, 4 * Hn);
  k_addvec<<<blks(4 * Hn), 256, 0, stream>>>(F(P_LSTM_BIH1), F(P_LSTM_BHH1), b_l1, 4 * Hn);
  k_addvec<<<blks(4 * HDn), 256, 0, stream>>>(F(P_DEC_BIH0), F(P_DEC_BHH0), b_d0, 4 * HDn);
  k_addvec<<<blks(4 * HDn), 256, 0, stream>>>(F(P_DEC_BIH1), F(P_DEC_BHH1), b_d1, 4 * HDn);
  float* dbias = ws.f32((long)Tact * 4 * HDn);
  k_dec_bias<<<blks((long)Tact * 4 * HDn), 256, 0, stream>>>(F(P_DEC_WIH0), dbias);

  // ---- activations --------------------------------------------------------
  half_t* X16 = ws.h16(Mseq * Fin);
  half_t* E1  = ws.h16(Mseq * 256);
  half_t* E2  = ws.h16(Mseq * 256);
  half_t* XG  = ws.h16(Mseq * 4L * Hn);     // reused: lstm0 then lstm1 input gates
  half_t* H0  = ws.h16(Mseq * Hn);
  half_t* H1  = ws.h16(Mseq * Hn);
  half_t* Gg  = ws.h16((long)Bsz * 4 * Hn); // step gate buffer (shared)
  half_t* h0b = ws.h16((long)Bsz * Hn);
  half_t* h1b = ws.h16((long)Bsz * Hn);
  float*  c0  = ws.f32((long)Bsz * Hn);
  float*  c1  = ws.f32((long)Bsz * Hn);
  half_t* XGc = ws.h16((long)Bsz * 4 * HDn);
  half_t* T2  = ws.h16((long)Bsz * 4 * HDn);
  half_t* DSq = ws.h16(Mdec * HDn);
  half_t* HH  = ws.h16(Mdec * 128);
  half_t* h1d = ws.h16((long)Bsz * HDn);
  half_t* h2d = ws.h16((long)Bsz * HDn);
  float*  c1d = ws.f32((long)Bsz * HDn);
  float*  c2d = ws.f32((long)Bsz * HDn);
  half_t* CHd = ws.h16((long)Bsz * 128);
  float*  CLg = ws.f32((long)Bsz * NCnt);

  // ---- encoder ------------------------------------------------------------
  k_f32_to_f16<<<blks(Mseq * Fin), 256, 0, stream>>>(F(IN_X), X16, Mseq * Fin);
  gemm(stream, X16, 0, Fin, w_enc1, F(P_ENC_B1), nullptr, 0, 0,
       E1, 0, 256, nullptr, 0, 0, (int)Mseq, 256, Fin, 1);
  gemm(stream, E1, 0, 256, w_enc2, F(P_ENC_B2), nullptr, 0, 0,
       E2, 0, 256, nullptr, 0, 0, (int)Mseq, 256, 256, 1);

  // ---- LSTM layer 0 -------------------------------------------------------
  gemm(stream, E2, 0, 256, w_ih0, b_l0, nullptr, 0, 0,
       XG, 0, 4 * Hn, nullptr, 0, 0, (int)Mseq, 4 * Hn, 256, 0);
  k_zero_h16<<<blks((long)Bsz * Hn), 256, 0, stream>>>(h0b, (long)Bsz * Hn);
  k_fill_f32<<<blks((long)Bsz * Hn), 256, 0, stream>>>(c0, 0.0f, (long)Bsz * Hn);
  for (int t = 0; t < Tseq; ++t) {
    gemm(stream, h0b, 0, Hn, w_hh0, nullptr, XG, (long)t * 4 * Hn, Tseq * 4 * Hn,
         Gg, 0, 4 * Hn, nullptr, 0, 0, Bsz, 4 * Hn, Hn, 0);
    k_lstm_cell<<<blks((long)Bsz * Hn), 256, 0, stream>>>(
        Gg, c0, h0b, H0, (long)t * Hn, Tseq * Hn, Bsz, Hn);
  }

  // ---- LSTM layer 1 -------------------------------------------------------
  gemm(stream, H0, 0, Hn, w_ih1, b_l1, nullptr, 0, 0,
       XG, 0, 4 * Hn, nullptr, 0, 0, (int)Mseq, 4 * Hn, Hn, 0);
  k_zero_h16<<<blks((long)Bsz * Hn), 256, 0, stream>>>(h1b, (long)Bsz * Hn);
  k_fill_f32<<<blks((long)Bsz * Hn), 256, 0, stream>>>(c1, 0.0f, (long)Bsz * Hn);
  for (int t = 0; t < Tseq; ++t) {
    gemm(stream, h1b, 0, Hn, w_hh1, nullptr, XG, (long)t * 4 * Hn, Tseq * 4 * Hn,
         Gg, 0, 4 * Hn, nullptr, 0, 0, Bsz, 4 * Hn, Hn, 0);
    k_lstm_cell<<<blks((long)Bsz * Hn), 256, 0, stream>>>(
        Gg, c1, h1b, H1, (long)t * Hn, Tseq * Hn, Bsz, Hn);
  }
  half_t* context = h1b;  // h at final timestep

  // ---- output offsets -----------------------------------------------------
  const long off_type = 0;
  const long off_mpos = off_type + Mdec * 4;
  const long off_clk  = off_mpos + Mdec * 2;
  const long off_key  = off_clk + Mdec * 3;
  const long off_scr  = off_key + Mdec * 50;
  const long off_tim  = off_scr + Mdec * 2;
  const long off_conf = off_tim + Mdec * 1;
  const long off_cnt  = off_conf + Mdec * 1;
  const long off_prob = off_cnt + Bsz;

  // ---- count head + softmax + action_count --------------------------------
  gemm(stream, context, 0, Hn, w_cnt1, F(P_CNT_B1), nullptr, 0, 0,
       CHd, 0, 128, nullptr, 0, 0, Bsz, 128, Hn, 1);
  gemm(stream, CHd, 0, 128, w_cnt2, F(P_CNT_B2), nullptr, 0, 0,
       nullptr, 0, 0, CLg, 0, NCnt, Bsz, NCnt, 128, 0);
  k_softmax16<<<blks(Bsz), 256, 0, stream>>>(CLg, dout + off_prob, Bsz);
  k_fill_f32<<<blks(Bsz), 256, 0, stream>>>(dout + off_cnt, (float)Tact, Bsz);

  // ---- decoder ------------------------------------------------------------
  gemm(stream, context, 0, Hn, w_dc, b_d0, nullptr, 0, 0,
       XGc, 0, 4 * HDn, nullptr, 0, 0, Bsz, 4 * HDn, Hn, 0);
  k_dec_init<<<blks((long)Bsz * HDn), 256, 0, stream>>>(context, h1d, h2d, c1d, c2d);
  for (int t = 0; t < Tact; ++t) {
    // layer 0: g1 = xg0[t] + h1 @ whh0^T   (xg0[t] = XGc + (t/T)*w_pos bias)
    gemm(stream, h1d, 0, HDn, w_dhh0, dbias + (long)t * 4 * HDn, XGc, 0, 4 * HDn,
         Gg, 0, 4 * HDn, nullptr, 0, 0, Bsz, 4 * HDn, HDn, 0);
    k_lstm_cell<<<blks((long)Bsz * HDn), 256, 0, stream>>>(
        Gg, c1d, h1d, nullptr, 0, 0, Bsz, HDn);
    // layer 1: g2 = h1 @ wih1^T + h2 @ whh1^T + b1g
    gemm(stream, h1d, 0, HDn, w_dih1, b_d1, nullptr, 0, 0,
         T2, 0, 4 * HDn, nullptr, 0, 0, Bsz, 4 * HDn, HDn, 0);
    gemm(stream, h2d, 0, HDn, w_dhh1, nullptr, T2, 0, 4 * HDn,
         Gg, 0, 4 * HDn, nullptr, 0, 0, Bsz, 4 * HDn, HDn, 0);
    k_lstm_cell<<<blks((long)Bsz * HDn), 256, 0, stream>>>(
        Gg, c2d, h2d, DSq, (long)t * HDn, Tact * HDn, Bsz, HDn);
  }

  // ---- output heads -------------------------------------------------------
  struct Head { int w1, b1, w2, b2, hid, nout, act; long off; };
  const Head heads[7] = {
      {P_TYPE_W1, P_TYPE_B1, P_TYPE_W2, P_TYPE_B2, 128, 4, 0, off_type},
      {P_MPOS_W1, P_MPOS_B1, P_MPOS_W2, P_MPOS_B2, 128, 2, 0, off_mpos},
      {P_CLK_W1,  P_CLK_B1,  P_CLK_W2,  P_CLK_B2,  128, 3, 0, off_clk},
      {P_KEY_W1,  P_KEY_B1,  P_KEY_W2,  P_KEY_B2,  128, 50, 0, off_key},
      {P_SCR_W1,  P_SCR_B1,  P_SCR_W2,  P_SCR_B2,  128, 2, 0, off_scr},
      {P_TIM_W1,  P_TIM_B1,  P_TIM_W2,  P_TIM_B2,  128, 1, 2, off_tim},
      {P_CONF_W1, P_CONF_B1, P_CONF_W2, P_CONF_B2, 64, 1, 2, off_conf},
  };
  for (int h = 0; h < 7; ++h) {
    const Head& hd = heads[h];
    half_t* hw1 = cvt(hd.w1, (long)hd.hid * HDn);
    half_t* hw2 = cvt(hd.w2, (long)hd.nout * hd.hid);
    gemm(stream, DSq, 0, HDn, hw1, F(hd.b1), nullptr, 0, 0,
         HH, 0, hd.hid, nullptr, 0, 0, (int)Mdec, hd.hid, HDn, 1);
    gemm(stream, HH, 0, hd.hid, hw2, F(hd.b2), nullptr, 0, 0,
         nullptr, 0, 0, dout + hd.off, 0, hd.nout, (int)Mdec, hd.nout, hd.hid, hd.act);
  }
}
